// LorentzMultiHeadAttention_36575941493347
// MI455X (gfx1250) — matrix operand order repaired and checked by reading, not compiled, not run on hardware
//
#include <hip/hip_runtime.h>
#include <hip/hip_bf16.h>

// ---------------------------------------------------------------------------
// Problem constants (match the reference)
// ---------------------------------------------------------------------------
#define BB   4
#define TT   2048
#define DMM  2048
#define HH   16
#define HKVV 4
#define DD   128

typedef __bf16 bf16;
typedef __attribute__((ext_vector_type(16))) __bf16 v16bf;
typedef __attribute__((ext_vector_type(8)))  __bf16 v8bf;
typedef __attribute__((ext_vector_type(4)))  __bf16 v4bf;
typedef __attribute__((ext_vector_type(8)))  float  v8f;

union BF16Frag { v16bf v; v8bf h[2]; };

static __device__ inline v16bf load_frag(const bf16* p0, const bf16* p1) {
    BF16Frag f;
    f.h[0] = *(const v8bf*)p0;
    f.h[1] = *(const v8bf*)p1;
    return f.v;
}

static __device__ inline v8f wmma_bf16(v16bf a, v16bf b, v8f c) {
    // v_wmma_f32_16x16x32_bf16  D = A(16x32) * B(32x16) + C
    return __builtin_amdgcn_wmma_f32_16x16x32_bf16(
        /*neg_a=*/false, a, /*neg_b=*/false, b,
        /*c_mod=*/(short)0, c, /*reuse_a=*/false, /*reuse_b=*/false);
}

// ---------------------------------------------------------------------------
// fp32 -> bf16 conversion (n is a multiple of 1024; 4 elems / thread)
// ---------------------------------------------------------------------------
__global__ void __launch_bounds__(256) cvt_f32_bf16(const float* __restrict__ in,
                                                    bf16* __restrict__ out, int n) {
    int i = (blockIdx.x * 256 + threadIdx.x) * 4;
    if (i >= n) return;
    float4 v = *(const float4*)(in + i);
    v4bf o;
    o.x = (bf16)v.x; o.y = (bf16)v.y; o.z = (bf16)v.z; o.w = (bf16)v.w;
    *(v4bf*)(out + i) = o;
}

// ---------------------------------------------------------------------------
// GEMM: C[M,N] = A[M,K] * W[N,K]^T   (A,W bf16)
// Block = 256 threads (8 waves), tile 128x128, BK=64 (two WMMA K-chunks/stage).
// Wave (wm,wn) computes a 32x64 sub-tile = 2x4 WMMA tiles; 16 wmma per stage.
// MODE 0: f32 row-major   MODE 1: bf16 row-major
// MODE 2: bf16, V-transposed store  [b][hkv][d][t]  (N = HKV*D, M = B*T)
// ---------------------------------------------------------------------------
template <int MODE>
__global__ void __launch_bounds__(256) gemm_bf16_wmma(const bf16* __restrict__ A,
                                                      const bf16* __restrict__ W,
                                                      void* __restrict__ Cout,
                                                      int M, int N, int K) {
    __shared__ bf16 As[128][80];   // stride 80 elems = 160B -> 16B-aligned frags
    __shared__ bf16 Bs[128][80];

    const int tid  = threadIdx.x;
    const int lane = tid & 31, wid = tid >> 5;
    const int wm   = wid & 3,  wn  = wid >> 2;     // 4 x 2 wave grid
    const int m0   = blockIdx.x * 128, n0 = blockIdx.y * 128;
    const int ln   = lane & 15, half = lane >> 4;
    const int srow = tid >> 1,  shalf = tid & 1;   // staging: 32 bf16 / thread / matrix

    v8f acc[2][4];
#pragma unroll
    for (int i = 0; i < 2; ++i)
#pragma unroll
        for (int j = 0; j < 4; ++j)
            acc[i][j] = (v8f){0.f,0.f,0.f,0.f,0.f,0.f,0.f,0.f};

    for (int kc = 0; kc < K; kc += 64) {
        // ---- stage 128x64 tiles of A and W into LDS ----
        {
            const bf16* ga = A + (size_t)(m0 + srow) * K + kc + 32 * shalf;
            const bf16* gw = W + (size_t)(n0 + srow) * K + kc + 32 * shalf;
#pragma unroll
            for (int u = 0; u < 4; ++u) {
                *(v8bf*)&As[srow][32 * shalf + 8 * u] = *(const v8bf*)(ga + 8 * u);
                *(v8bf*)&Bs[srow][32 * shalf + 8 * u] = *(const v8bf*)(gw + 8 * u);
            }
        }
        __syncthreads();

#pragma unroll
        for (int c = 0; c < 2; ++c) {
            // A frags (contiguous k order: lanes<16 K0..15, lanes>=16 K16..31)
            v16bf af[2];
#pragma unroll
            for (int i = 0; i < 2; ++i) {
                const bf16* p = &As[wm * 32 + i * 16 + ln][32 * c];
                af[i] = load_frag(p + 16 * half, p + 16 * half + 8);
            }
            // B frags (B[k][n] = W[n][k]; lanes<16 K0..15, lanes>=16 K16..31)
            v16bf bfg[4];
#pragma unroll
            for (int j = 0; j < 4; ++j) {
                const bf16* p = &Bs[wn * 64 + j * 16 + ln][32 * c];
                bfg[j] = load_frag(p + 16 * half, p + 16 * half + 8);
            }
#pragma unroll
            for (int i = 0; i < 2; ++i)
#pragma unroll
                for (int j = 0; j < 4; ++j)
                    acc[i][j] = wmma_bf16(af[i], bfg[j], acc[i][j]);
        }
        __syncthreads();
    }

    // ---- epilogue (C layout: lanes<16 -> rows r, lanes>=16 -> rows r+8; col = ln)
#pragma unroll
    for (int i = 0; i < 2; ++i)
#pragma unroll
        for (int j = 0; j < 4; ++j)
#pragma unroll
            for (int r = 0; r < 8; ++r) {
                int row = m0 + wm * 32 + i * 16 + r + 8 * half;
                int col = n0 + wn * 64 + j * 16 + ln;
                float v = acc[i][j][r];
                if (MODE == 0) {
                    ((float*)Cout)[(size_t)row * N + col] = v;
                } else if (MODE == 1) {
                    ((bf16*)Cout)[(size_t)row * N + col] = (bf16)v;
                } else {  // MODE 2: V transposed [b][hkv][d][t]
                    int b = row >> 11, t = row & (TT - 1);
                    int hk = col >> 7, d = col & (DD - 1);
                    ((bf16*)Cout)[(((size_t)b * HKVV + hk) * DD + d) * TT + t] = (bf16)v;
                }
            }
}

// ---------------------------------------------------------------------------
// RMS-norm + RoPE, in place on bf16 q [B,T,H,D] and k [B,T,HKV,D].
// One wave per 128-dim head vector; lane holds d = lane, lane+32, lane+64, lane+96
// so RoPE pairs (i, i+64) stay inside one lane.
// The attention scale 1/sqrt(D) is folded into q here.
// ---------------------------------------------------------------------------
__global__ void __launch_bounds__(256) norm_rope_kernel(bf16* __restrict__ q,
                                                        bf16* __restrict__ k) {
    const int lane = threadIdx.x & 31, wid = threadIdx.x >> 5;
    const long vid = (long)blockIdx.x * 8 + wid;
    const long NQ = (long)BB * TT * HH;

    bf16* p;
    int t;
    float post = 1.0f;
    if (vid < NQ) {
        long row = vid / HH; int h = (int)(vid % HH);
        t = (int)(row % TT);
        p = q + row * (HH * DD) + h * DD;
        post = 0.08838834764831845f;   // 1/sqrt(128) folded into q
    } else {
        long v2 = vid - NQ;
        long row = v2 / HKVV; int h = (int)(v2 % HKVV);
        t = (int)(row % TT);
        p = k + row * (HKVV * DD) + h * DD;
    }

    float e[4];
#pragma unroll
    for (int j = 0; j < 4; ++j) e[j] = (float)p[lane + 32 * j];

    float ss = e[0]*e[0] + e[1]*e[1] + e[2]*e[2] + e[3]*e[3];
#pragma unroll
    for (int m = 1; m < 32; m <<= 1) ss += __shfl_xor(ss, m, 32);
    float rn = rsqrtf(ss * (1.f / 128.f) + 1.1920929e-7f);
#pragma unroll
    for (int j = 0; j < 4; ++j) e[j] *= rn;

    // inv_freq[i] = 10000^(-i/64), pairs i = lane and lane+32
    const float NLOG = -9.210340371976184f / 64.f;  // -ln(10000)/64
    float f0 = (float)t * __expf((float)lane * NLOG);
    float f1 = (float)t * __expf((float)(lane + 32) * NLOG);
    float s0, c0, s1, c1;
    __sincosf(f0, &s0, &c0);
    __sincosf(f1, &s1, &c1);

    float o0 = ( e[0] * c0 + e[2] * s0) * post;
    float o2 = (-e[0] * s0 + e[2] * c0) * post;
    float o1 = ( e[1] * c1 + e[3] * s1) * post;
    float o3 = (-e[1] * s1 + e[3] * c1) * post;

    p[lane]      = (bf16)o0;
    p[lane + 32] = (bf16)o1;
    p[lane + 64] = (bf16)o2;
    p[lane + 96] = (bf16)o3;
}

// ---------------------------------------------------------------------------
// Causal flash attention (GQA). Grid (B*H, T/64), block 128 (4 waves).
// Each wave owns one 16-query tile; keys processed 32 at a time:
//   S(16x32) via 8 WMMA, static-max streaming softmax, P via LDS, PV via 8 WMMA.
// Scores are bounded: |s| = |qhat.khat|*sqrt(D) <= ~11.3 (rms-normed q,k with
// 1/sqrt(D) folded into q), so exp(s - 12) cannot overflow -> no running max,
// no accumulator rescale.  Row sums are kept as per-lane partials and reduced
// once at the end.  Only the final 32-key block can cross the causal diagonal,
// so the main loop is mask-free.
// Q [B,T,H,D] bf16 (pre-scaled), K [B,T,HKV,D] bf16, V transposed [B,HKV,D,T].
// Output attn [B,T,H*D] bf16.
// ---------------------------------------------------------------------------
__global__ void __launch_bounds__(128) attn_fa_wmma(const bf16* __restrict__ Q,
                                                    const bf16* __restrict__ Kr,
                                                    const bf16* __restrict__ Vt,
                                                    bf16* __restrict__ Oa) {
    __shared__ bf16 Plds[4][16][32];   // per-wave P staging (A-layout reload)

    const int lane = threadIdx.x & 31, wid = threadIdx.x >> 5;
    const int b = blockIdx.x / HH, h = blockIdx.x % HH, hkv = h >> 2;
    const int q0 = (blockIdx.y * 4 + wid) * 16;
    const int ln = lane & 15, half = lane >> 4;
    const float MSTAT = 12.0f;         // static softmax max bound

    float lrow[8];                     // per-lane PARTIAL row sums
    v8f o[8];
#pragma unroll
    for (int r = 0; r < 8; ++r) lrow[r] = 0.f;
#pragma unroll
    for (int f = 0; f < 8; ++f) o[f] = (v8f){0.f,0.f,0.f,0.f,0.f,0.f,0.f,0.f};

    // Loop-invariant Q A-fragments (row M = ln), 4 K-chunks of 32 features.
    const bf16* qp = Q + ((size_t)(b * TT + q0 + ln)) * (HH * DD) + h * DD;
    v16bf qa[4];
#pragma unroll
    for (int c = 0; c < 4; ++c)
        qa[c] = load_frag(qp + 32 * c + 16 * half, qp + 32 * c + 16 * half + 8);

    auto block_step = [&](int kb, bool mask) {
        // ---- S = Q*K^T for 32 keys (two 16x16 tiles, 8 WMMA) ----
        v16bf bk[8];
#pragma unroll
        for (int c = 0; c < 4; ++c) {
            const bf16* kp0 = Kr + (size_t)(b * TT + kb + ln) * (HKVV * DD)
                              + hkv * DD + 32 * c + 16 * half;
            const bf16* kp1 = kp0 + (size_t)16 * (HKVV * DD);
            bk[2 * c]     = load_frag(kp0, kp0 + 8);
            bk[2 * c + 1] = load_frag(kp1, kp1 + 8);
        }
        v8f s0 = (v8f){0.f,0.f,0.f,0.f,0.f,0.f,0.f,0.f};
        v8f s1 = s0;
#pragma unroll
        for (int c = 0; c < 4; ++c) {
            s0 = wmma_bf16(qa[c], bk[2 * c],     s0);
            s1 = wmma_bf16(qa[c], bk[2 * c + 1], s1);
        }

        // ---- streaming softmax with static max ----
        const int qbase = q0 + 8 * half;
        const int k0 = kb + ln, k1 = kb + 16 + ln;
#pragma unroll
        for (int r = 0; r < 8; ++r) {
            float p0 = __expf(s0[r] - MSTAT);
            float p1 = __expf(s1[r] - MSTAT);
            if (mask) {
                if (k0 > qbase + r) p0 = 0.f;
                if (k1 > qbase + r) p1 = 0.f;
            }
            lrow[r] += p0 + p1;
            Plds[wid][8 * half + r][ln]      = (bf16)p0;
            Plds[wid][8 * half + r][16 + ln] = (bf16)p1;
        }
        asm volatile("s_wait_dscnt 0" ::: "memory");  // same-wave LDS RAW

        // ---- O += P * V (A = P 16x32 from LDS, B = V from transposed global) ----
        const bf16* pp = &Plds[wid][ln][0];
        v16bf pa = load_frag(pp + 16 * half, pp + 16 * half + 8);
        v16bf vb[8];
#pragma unroll
        for (int f = 0; f < 8; ++f) {
            const bf16* vp = Vt + ((size_t)(b * HKVV + hkv) * DD + f * 16 + ln) * TT
                             + kb + 16 * half;
            vb[f] = load_frag(vp, vp + 8);
        }
#pragma unroll
        for (int f = 0; f < 8; ++f)
            o[f] = wmma_bf16(pa, vb[f], o[f]);
    };

    const int kend = q0 + 16;
    const int kb_last = ((kend - 1) >> 5) << 5;   // only block crossing diagonal
    for (int kb = 0; kb < kb_last; kb += 32)
        block_step(kb, false);
    block_step(kb_last, true);

    // ---- final row-sum reduction (once), normalize, store ----
    float inv[8];
#pragma unroll
    for (int r = 0; r < 8; ++r) {
        float s = lrow[r];
#pragma unroll
        for (int m = 1; m < 16; m <<= 1) s += __shfl_xor(s, m, 32);
        inv[r] = 1.0f / s;
    }
#pragma unroll
    for (int f = 0; f < 8; ++f)
#pragma unroll
        for (int r = 0; r < 8; ++r) {
            float val = o[f][r] * inv[r];
            size_t row = (size_t)b * TT + q0 + 8 * half + r;
            Oa[row * (HH * DD) + h * DD + f * 16 + ln] = (bf16)val;
        }
}

__global__ void set_scalar_zero(float* p) { *p = 0.f; }

// ---------------------------------------------------------------------------
// Host launcher
// ---------------------------------------------------------------------------
extern "C" void kernel_launch(void* const* d_in, const int* in_sizes, int n_in,
                              void* d_out, int out_size, void* d_ws, size_t ws_size,
                              hipStream_t stream) {
    (void)in_sizes; (void)n_in; (void)out_size; (void)ws_size;
    const float* x  = (const float*)d_in[0];
    const float* Wq = (const float*)d_in[1];
    const float* Wk = (const float*)d_in[2];
    const float* Wv = (const float*)d_in[3];
    const float* Wo = (const float*)d_in[4];
    float* out = (float*)d_out;

    const size_t M = (size_t)BB * TT;  // 8192 rows

    char* ws = (char*)d_ws;
    size_t off = 0;
    auto alloc = [&](size_t bytes) -> bf16* {
        bf16* p = (bf16*)(ws + off);
        off += (bytes + 255) & ~(size_t)255;
        return p;
    };
    bf16* xb  = alloc(M * DMM * 2);                 // x bf16
    bf16* Wqb = alloc((size_t)HH * DD * DMM * 2);   // Wq bf16
    bf16* Wkb = alloc((size_t)HKVV * DD * DMM * 2); // Wk bf16
    bf16* Wvb = alloc((size_t)HKVV * DD * DMM * 2); // Wv bf16
    bf16* Wob = alloc((size_t)DMM * HH * DD * 2);   // Wo bf16
    bf16* qb  = alloc(M * HH * DD * 2);             // q [B,T,H,D]
    bf16* kb  = alloc(M * HKVV * DD * 2);           // k [B,T,HKV,D]
    bf16* vtb = alloc(M * HKVV * DD * 2);           // v transposed [B,HKV,D,T]
    bf16* ab  = alloc(M * HH * DD * 2);             // attention output bf16

    auto cvt = [&](const float* in, bf16* o, size_t n) {
        cvt_f32_bf16<<<dim3((unsigned)(n / 1024)), dim3(256), 0, stream>>>(in, o, (int)n);
    };
    cvt(x,  xb,  M * DMM);
    cvt(Wq, Wqb, (size_t)HH * DD * DMM);
    cvt(Wk, Wkb, (size_t)HKVV * DD * DMM);
    cvt(Wv, Wvb, (size_t)HKVV * DD * DMM);
    cvt(Wo, Wob, (size_t)DMM * HH * DD);

    // QKV projections (WMMA GEMMs)
    gemm_bf16_wmma<1><<<dim3(64, 16), 256, 0, stream>>>(xb, Wqb, qb, 8192, 2048, 2048);
    gemm_bf16_wmma<1><<<dim3(64, 4),  256, 0, stream>>>(xb, Wkb, kb, 8192, 512, 2048);
    gemm_bf16_wmma<2><<<dim3(64, 4),  256, 0, stream>>>(xb, Wvb, vtb, 8192, 512, 2048);

    // RMS-norm + RoPE in place on q, k (q also picks up the 1/sqrt(D) scale)
    norm_rope_kernel<<<dim3(20480), 256, 0, stream>>>(qb, kb);

    // Flash attention
    attn_fa_wmma<<<dim3(64, 32), 128, 0, stream>>>(qb, kb, vtb, ab);

    // Output projection -> fp32 d_out
    gemm_bf16_wmma<0><<<dim3(64, 16), 256, 0, stream>>>(ab, Wob, out, 8192, 2048, 2048);

    // spatial_norm = 0
    set_scalar_zero<<<1, 1, 0, stream>>>(out + M * DMM);
}
